// GNN_ECODQN_67405216744188
// MI455X (gfx1250) — matrix operand order repaired and checked by reading, not compiled, not run on hardware
//
#include <hip/hip_runtime.h>
#include <hip/hip_bf16.h>

typedef __attribute__((ext_vector_type(16))) _Float16 v16h;
typedef __attribute__((ext_vector_type(8)))  float    v8f;

constexpr int N   = 50000;
constexpr int DIN = 7;
constexpr int E   = 800000;
constexpr int D   = 64;
constexpr int L   = 3;
constexpr int G   = 64;
constexpr int NP  = 50048;   // N padded to 64-row GEMM tiles (782 blocks * 64)

union FragH { v16h h; unsigned u[8]; };

// ---------------------------------------------------------------- utilities
__global__ void zero_f32_kernel(float* __restrict__ p, int n) {
  int i = blockIdx.x * blockDim.x + threadIdx.x;
  if (i < n) p[i] = 0.0f;
}

// Swizzle f32 weight matrix W[K=KC*32, 64] into B-fragment order:
// dst[((nt*KC + kc)*32 + lane)*8 + v] packs halfs B[k][n], B[k+1][n]
// with n = nt*16 + lane%16, k = kc*32 + (v/4)*16 + (lane/16)*8 + 2*(v%4).
__global__ void swz_kernel(const float* __restrict__ W, unsigned* __restrict__ dst, int KC) {
  int idx = blockIdx.x * blockDim.x + threadIdx.x;
  int total = 4 * KC * 32 * 8;
  if (idx >= total) return;
  int v    = idx & 7;
  int lane = (idx >> 3) & 31;
  int rest = idx >> 8;
  int kc   = rest % KC;
  int nt   = rest / KC;
  int n = nt * 16 + (lane & 15);
  int k = kc * 32 + (v >> 2) * 16 + ((lane >> 4) & 1) * 8 + 2 * (v & 3);
  union { _Float16 h[2]; unsigned u; } p;
  p.h[0] = (_Float16)W[(k + 0) * D + n];
  p.h[1] = (_Float16)W[(k + 1) * D + n];
  dst[idx] = p.u;
}

// ------------------------------------------------------- WMMA GEMM (core)
// D = relu(A[N,KC*32] @ B[KC*32,64] + bias).  A is f16 row-major (as u32),
// B pre-swizzled.  Optional f32 output [N,64] and/or f16 output (stride/off).
template<int KC>
__global__ __launch_bounds__(128)
void wmma_gemm_kernel(const unsigned* __restrict__ A,
                      const unsigned* __restrict__ Bswz,
                      const float* __restrict__ bias,
                      float* __restrict__ outF,
                      _Float16* __restrict__ outH,
                      int outHStride, int outHOff)
{
  const int lane    = threadIdx.x & 31;
  const int wave    = threadIdx.x >> 5;
  const int rowBase = blockIdx.x * 64 + wave * 16;
  const int rowDw   = KC * 16;            // dwords per A row
  const int hi      = (lane >> 4) & 1;

  v8f acc[4];
  v8f z = {};
#pragma unroll
  for (int i = 0; i < 4; ++i) acc[i] = z;

  const int aRow = rowBase + (lane & 15);
  const unsigned* aPtr = A + (size_t)aRow * rowDw + hi * 4;

#pragma unroll
  for (int kc = 0; kc < KC; ++kc) {
    FragH af;
#pragma unroll
    for (int v = 0; v < 8; ++v)
      af.u[v] = aPtr[kc * 16 + (v >> 2) * 8 + (v & 3)];   // two b128 loads
#pragma unroll
    for (int nt = 0; nt < 4; ++nt) {
      FragH bf;
#pragma unroll
      for (int v = 0; v < 8; ++v)
        bf.u[v] = Bswz[((nt * KC + kc) * 32 + lane) * 8 + v];  // coalesced b128
      acc[nt] = __builtin_amdgcn_wmma_f32_16x16x32_f16(
          false, af.h, false, bf.h, (short)0, acc[nt], false, false);
    }
  }

  const int mBase = rowBase + hi * 8;
#pragma unroll
  for (int nt = 0; nt < 4; ++nt) {
    const int col = nt * 16 + (lane & 15);
    const float bb = bias[col];
#pragma unroll
    for (int r = 0; r < 8; ++r) {
      const int row = mBase + r;
      if (row < N) {
        float val = fmaxf(acc[nt][r] + bb, 0.0f);
        if (outF) outF[(size_t)row * D + col] = val;
        if (outH) outH[(size_t)row * outHStride + outHOff + col] = (_Float16)val;
      }
    }
  }
}

// ---------------------------------------------------------- pipeline stages
__global__ void encoder_kernel(const float* __restrict__ x, const float* __restrict__ W,
                               const float* __restrict__ b, float* __restrict__ xemb,
                               _Float16* __restrict__ Au) {
  int idx = blockIdx.x * blockDim.x + threadIdx.x;
  if (idx >= N * D) return;
  int n = idx >> 6, d = idx & 63;
  float a = b[d];
#pragma unroll
  for (int j = 0; j < DIN; ++j) a += x[n * DIN + j] * W[j * D + d];
  a = fmaxf(a, 0.0f);
  xemb[idx] = a;
  Au[(size_t)n * 128 + d] = (_Float16)a;
}

__global__ void edge_count_kernel(const int* __restrict__ ei, float* __restrict__ cnt) {
  int e = blockIdx.x * blockDim.x + threadIdx.x;
  if (e >= E) return;
  atomicAdd(&cnt[ei[E + e]], 1.0f);
}

__global__ void ne_scatter_kernel(const int* __restrict__ ei, const float* __restrict__ x,
                                  const float* __restrict__ ea, float* __restrict__ ne_acc) {
  int idx = blockIdx.x * blockDim.x + threadIdx.x;
  if (idx >= E * 8) return;
  int e = idx >> 3, d = idx & 7;
  int col = ei[e], row = ei[E + e];
  float v = (d < DIN) ? x[col * DIN + d] : ea[e];
  atomicAdd(&ne_acc[row * 8 + d], v);
}

__global__ void degmax_kernel(const int* __restrict__ batch, const float* __restrict__ degree,
                              float* __restrict__ degmax) {
  int n = blockIdx.x * blockDim.x + threadIdx.x;
  if (n >= N) return;
  atomicMax((int*)&degmax[batch[n]], __float_as_int(degree[n]));  // degrees >= 0
}

__global__ void ne_finalize_kernel(const float* __restrict__ ne_acc, const float* __restrict__ cnt,
                                   const int* __restrict__ batch, const float* __restrict__ degmax,
                                   const float* __restrict__ W, const float* __restrict__ b,
                                   _Float16* __restrict__ Ag) {
  int idx = blockIdx.x * blockDim.x + threadIdx.x;
  if (idx >= N * D) return;
  int n = idx >> 6, d = idx & 63;
  float v;
  if (d == D - 1) {
    v = degmax[batch[n]];                       // degree_norm channel (no relu)
  } else {
    float inv = 1.0f / fmaxf(cnt[n], 1.0f);
    float a = b[d];
#pragma unroll
    for (int j = 0; j < DIN + 1; ++j) a += ne_acc[n * 8 + j] * inv * W[j * (D - 1) + d];
    v = fmaxf(a, 0.0f);
  }
  Ag[(size_t)n * D + d] = (_Float16)v;
}

__global__ void agg_scatter_kernel(const int* __restrict__ ei, const float* __restrict__ ea,
                                   const float* __restrict__ xemb, float* __restrict__ acc) {
  int idx = blockIdx.x * blockDim.x + threadIdx.x;
  if (idx >= E * D) return;
  int e = idx >> 6, d = idx & 63;
  int col = ei[e], row = ei[E + e];
  atomicAdd(&acc[row * D + d], ea[e] * xemb[col * D + d]);
}

__global__ void agg_finalize_kernel(const float* __restrict__ acc, const float* __restrict__ cnt,
                                    _Float16* __restrict__ Am) {
  int idx = blockIdx.x * blockDim.x + threadIdx.x;
  if (idx >= N * D) return;
  int n = idx >> 6, d = idx & 63;
  Am[(size_t)n * 128 + d] = (_Float16)(acc[idx] / fmaxf(cnt[n], 1.0f));
}

__global__ void g_scatter_kernel(const int* __restrict__ batch, const float* __restrict__ xemb,
                                 float* __restrict__ gacc, float* __restrict__ gcnt) {
  int idx = blockIdx.x * blockDim.x + threadIdx.x;
  if (idx >= N * D) return;
  int n = idx >> 6, d = idx & 63;
  int b = batch[n];
  atomicAdd(&gacc[b * D + d], xemb[idx]);
  if (d == 0) atomicAdd(&gcnt[b], 1.0f);
}

__global__ void g2_kernel(const float* __restrict__ gacc, const float* __restrict__ gcnt,
                          const float* __restrict__ W, const float* __restrict__ b,
                          float* __restrict__ g2) {
  int g = blockIdx.x, d = threadIdx.x;
  float inv = 1.0f / fmaxf(gcnt[g], 1.0f);
  float a = b[d];
  for (int j = 0; j < D; ++j) a += gacc[g * D + j] * inv * W[j * D + d];
  g2[g * D + d] = a;   // reference applies no relu to g_agg @ W_g + b_g
}

__global__ void readout_kernel(const int* __restrict__ batch, const float* __restrict__ g2,
                               const float* __restrict__ xemb, const float* __restrict__ W_r,
                               const float* __restrict__ b_r, float* __restrict__ out) {
  int n = blockIdx.x * blockDim.x + threadIdx.x;
  if (n >= N) return;
  int b = batch[n];
  float q = b_r[0];
#pragma unroll 8
  for (int j = 0; j < D; ++j) q += fmaxf(g2[b * D + j], 0.0f) * W_r[j];
#pragma unroll 8
  for (int j = 0; j < D; ++j) q += fmaxf(xemb[(size_t)n * D + j], 0.0f) * W_r[D + j];
  out[n] = q;
}

// ------------------------------------------------------------------- driver
extern "C" void kernel_launch(void* const* d_in, const int* in_sizes, int n_in,
                              void* d_out, int out_size, void* d_ws, size_t ws_size,
                              hipStream_t stream) {
  (void)in_sizes; (void)n_in; (void)out_size; (void)ws_size;
  const float* x      = (const float*)d_in[0];
  const int*   ei     = (const int*)d_in[1];     // [2,E]
  const float* ea     = (const float*)d_in[2];
  const int*   batch  = (const int*)d_in[3];
  const float* degree = (const float*)d_in[4];
  const float* W_en   = (const float*)d_in[6];
  const float* b_en   = (const float*)d_in[7];
  const float* W_ene  = (const float*)d_in[8];
  const float* b_ene  = (const float*)d_in[9];
  const float* W_agg  = (const float*)d_in[10];
  const float* b_agg  = (const float*)d_in[11];
  const float* Wm     = (const float*)d_in[12];
  const float* bm     = (const float*)d_in[13];
  const float* Wu     = (const float*)d_in[14];
  const float* bu     = (const float*)d_in[15];
  const float* W_g    = (const float*)d_in[16];
  const float* b_g    = (const float*)d_in[17];
  const float* W_r    = (const float*)d_in[18];
  const float* b_r    = (const float*)d_in[19];
  float* out = (float*)d_out;

  char* ws = (char*)d_ws;
  size_t off = 0;
  auto alloc = [&](size_t bytes) -> char* {
    char* p = ws + off;
    off = (off + bytes + 255) & ~(size_t)255;
    return p;
  };
  float*    cnt     = (float*)alloc((size_t)NP * sizeof(float));
  float*    ne_acc  = (float*)alloc((size_t)NP * 8 * sizeof(float));
  float*    agg_acc = (float*)alloc((size_t)NP * D * sizeof(float));
  float*    xemb    = (float*)alloc((size_t)NP * D * sizeof(float));
  float*    degmax  = (float*)alloc(G * sizeof(float));
  float*    gacc    = (float*)alloc(G * D * sizeof(float));
  float*    gcnt    = (float*)alloc(G * sizeof(float));
  float*    g2      = (float*)alloc(G * D * sizeof(float));
  unsigned* WaggS   = (unsigned*)alloc(4 * 2 * 32 * 8 * sizeof(unsigned));
  unsigned* WmS     = (unsigned*)alloc((size_t)L * 4096 * sizeof(unsigned));
  unsigned* WuS     = (unsigned*)alloc((size_t)L * 4096 * sizeof(unsigned));
  _Float16* Ag      = (_Float16*)alloc((size_t)NP * D * sizeof(_Float16));
  _Float16* Am      = (_Float16*)alloc((size_t)NP * 128 * sizeof(_Float16));
  _Float16* Au      = (_Float16*)alloc((size_t)NP * 128 * sizeof(_Float16));

  auto cdiv = [](int a, int b) { return (a + b - 1) / b; };
  const int TB = 256;

  // init accumulators (ws is not re-zeroed between calls)
  zero_f32_kernel<<<cdiv(NP, TB), TB, 0, stream>>>(cnt, NP);
  zero_f32_kernel<<<cdiv(NP * 8, TB), TB, 0, stream>>>(ne_acc, NP * 8);
  zero_f32_kernel<<<1, 64, 0, stream>>>(degmax, G);
  zero_f32_kernel<<<cdiv(G * D, TB), TB, 0, stream>>>(gacc, G * D);
  zero_f32_kernel<<<1, 64, 0, stream>>>(gcnt, G);

  // weight swizzle (f32 -> f16 B-fragment order)
  swz_kernel<<<cdiv(4 * 2 * 32 * 8, TB), TB, 0, stream>>>(W_agg, WaggS, 2);
  for (int l = 0; l < L; ++l) {
    swz_kernel<<<cdiv(4096, TB), TB, 0, stream>>>(Wm + (size_t)l * 128 * D, WmS + l * 4096, 4);
    swz_kernel<<<cdiv(4096, TB), TB, 0, stream>>>(Wu + (size_t)l * 128 * D, WuS + l * 4096, 4);
  }

  // encoder + edge-feature pipeline
  encoder_kernel<<<cdiv(N * D, TB), TB, 0, stream>>>(x, W_en, b_en, xemb, Au);
  edge_count_kernel<<<cdiv(E, TB), TB, 0, stream>>>(ei, cnt);
  ne_scatter_kernel<<<cdiv(E * 8, TB), TB, 0, stream>>>(ei, x, ea, ne_acc);
  degmax_kernel<<<cdiv(N, TB), TB, 0, stream>>>(batch, degree, degmax);
  ne_finalize_kernel<<<cdiv(N * D, TB), TB, 0, stream>>>(ne_acc, cnt, batch, degmax,
                                                         W_ene, b_ene, Ag);
  // x_agg_emb = relu(Ag @ W_agg + b_agg) -> right half of Am (f16)
  wmma_gemm_kernel<2><<<NP / 64, 128, 0, stream>>>((const unsigned*)Ag, WaggS, b_agg,
                                                   nullptr, Am, 128, 64);

  // message-passing layers
  for (int l = 0; l < L; ++l) {
    zero_f32_kernel<<<cdiv(NP * D, TB), TB, 0, stream>>>(agg_acc, NP * D);
    agg_scatter_kernel<<<cdiv(E * D, TB), TB, 0, stream>>>(ei, ea, xemb, agg_acc);
    agg_finalize_kernel<<<cdiv(N * D, TB), TB, 0, stream>>>(agg_acc, cnt, Am);
    // m = relu([x_agg | x_agg_emb] @ Wm + bm) -> right half of Au
    wmma_gemm_kernel<4><<<NP / 64, 128, 0, stream>>>((const unsigned*)Am, WmS + l * 4096,
                                                     bm + l * D, nullptr, Au, 128, 64);
    // x_emb = relu([x_emb | m] @ Wu + bu) -> f32 xemb + left half of Au
    wmma_gemm_kernel<4><<<NP / 64, 128, 0, stream>>>((const unsigned*)Au, WuS + l * 4096,
                                                     bu + l * D, xemb, Au, 128, 0);
  }

  // graph pooling + readout
  g_scatter_kernel<<<cdiv(N * D, TB), TB, 0, stream>>>(batch, xemb, gacc, gcnt);
  g2_kernel<<<G, D, 0, stream>>>(gacc, gcnt, W_g, b_g, g2);
  readout_kernel<<<cdiv(N, TB), TB, 0, stream>>>(batch, g2, xemb, W_r, b_r, out);
}
